// HeteroGNN_51788715655346
// MI455X (gfx1250) — matrix operand order repaired and checked by reading, not compiled, not run on hardware
//
#include <hip/hip_runtime.h>

#define N_NODES 100000
#define HDIM 64
#define NREL 14
#define NEDGE 100000
#define NPF 800000

typedef __attribute__((ext_vector_type(16))) __bf16 v16bf;
typedef __attribute__((ext_vector_type(8)))  __bf16 v8bf;
typedef __attribute__((ext_vector_type(8)))  float  v8f;

// ---------------- helpers ----------------

__device__ inline unsigned short f2bf(float f) {
    unsigned int u = __float_as_uint(f);
    unsigned int r = u + 0x7FFFu + ((u >> 16) & 1u);   // round-to-nearest-even
    return (unsigned short)(r >> 16);
}

__device__ inline void atomicMaxFloat(float* addr, float val) {
    if (val >= 0.0f) atomicMax((int*)addr, __float_as_int(val));
    else             atomicMin((unsigned int*)addr, (unsigned int)__float_as_int(val));
}

union FragU { v16bf v; v8bf h[2]; };

// A-fragment: 16x32 bf16, row = lane&15. Per ISA layout:
// lanes 0-15: elems 0..7 -> K = ks*32 + 0..7,  elems 8..15 -> K = ks*32 + 16..23
// lanes 16-31: elems 0..7 -> K = ks*32 + 8..15, elems 8..15 -> K = ks*32 + 24..31
__device__ inline v16bf load_fragA(const unsigned short* rowPtr, int kstep) {
    int lane = threadIdx.x & 31;
    int kb = (lane >> 4) * 8;
    FragU u;
    u.h[0] = *reinterpret_cast<const v8bf*>(rowPtr + kstep * 32 + kb);
    u.h[1] = *reinterpret_cast<const v8bf*>(rowPtr + kstep * 32 + 16 + kb);
    return u.v;
}

// B-fragment: 32x16 bf16 from transposed weights wT[col][dinPad].
// lane&15 = column; lanes 0-15 hold K=0..7,16..23, lanes 16-31 hold K=8..15,24..31 (matches A).
__device__ inline v16bf load_fragB(const unsigned short* wT, int dinPad, int colTile, int kstep) {
    int lane = threadIdx.x & 31;
    int col = colTile * 16 + (lane & 15);
    int kb = (lane >> 4) * 8;
    const unsigned short* p = wT + (size_t)col * dinPad + kstep * 32 + kb;
    FragU u;
    u.h[0] = *reinterpret_cast<const v8bf*>(p);
    u.h[1] = *reinterpret_cast<const v8bf*>(p + 16);
    return u.v;
}

// ---------------- conversion kernels ----------------

__global__ void zero_kernel(float* __restrict__ p, int n) {
    int i = blockIdx.x * blockDim.x + threadIdx.x;
    if (i < n) p[i] = 0.0f;
}

// x[N][4] -> hbf[N][32] bf16, zero padded
__global__ void cvt_x_kernel(const float* __restrict__ x, unsigned short* __restrict__ hbf) {
    int i = blockIdx.x * blockDim.x + threadIdx.x;
    if (i < N_NODES * 32) {
        int row = i >> 5, k = i & 31;
        hbf[i] = (k < 4) ? f2bf(x[row * 4 + k]) : (unsigned short)0;
    }
}

// relu(h_acc[N][64]) -> hbf[N][64] bf16
__global__ void relu_cvt_kernel(const float* __restrict__ h, unsigned short* __restrict__ hbf) {
    int i = blockIdx.x * blockDim.x + threadIdx.x;
    if (i < N_NODES * HDIM) {
        float v = h[i];
        hbf[i] = f2bf(v > 0.0f ? v : 0.0f);
    }
}

// W[NREL][din][64] -> dst[NREL][64][dinPad] bf16 transposed, zero K-pad
__global__ void cvtWT_kernel(const float* __restrict__ W, unsigned short* __restrict__ dst,
                             int din, int dinPad) {
    int i = blockIdx.x * blockDim.x + threadIdx.x;
    int total = NREL * HDIM * dinPad;
    if (i < total) {
        int r   = i / (HDIM * dinPad);
        int rem = i % (HDIM * dinPad);
        int col = rem / dinPad;
        int k   = rem % dinPad;
        dst[i] = (k < din) ? f2bf(W[((size_t)r * din + k) * HDIM + col]) : (unsigned short)0;
    }
}

__global__ void seg_init_kernel(float* __restrict__ dmax, float* __restrict__ dsum,
                                int dlo, int cnt) {
    int i = blockIdx.x * blockDim.x + threadIdx.x;
    if (i < cnt) {
        dmax[dlo + i] = __int_as_float(0xFF800000);  // -inf
        dsum[dlo + i] = 0.0f;
    }
}

// ---------------- WMMA GEMM: Y = X@W + b ----------------
// MODE 0: out[row_rel*64+col] = acc + bias (qbuf)
// MODE 1: out[(rowBase+row)*64+col] += acc + bias (self term into h_acc)

template <int KSTEPS, int MODE>
__global__ void gemm_bias_kernel(const unsigned short* __restrict__ hbf, int dinPad,
                                 const unsigned short* __restrict__ WT,
                                 const float* __restrict__ bias,
                                 float* __restrict__ out, int rowBase, int M) {
    int wave = blockIdx.x * (blockDim.x >> 5) + (threadIdx.x >> 5);
    int rows = wave * 16;
    if (rows >= M) return;
    int lane = threadIdx.x & 31;
    int node = rowBase + rows + (lane & 15);
    const unsigned short* rowPtr = hbf + (size_t)node * dinPad;

    v16bf a[KSTEPS];
#pragma unroll
    for (int ks = 0; ks < KSTEPS; ++ks) a[ks] = load_fragA(rowPtr, ks);
    int mrow = (lane >> 4) * 8;

#pragma unroll
    for (int ct = 0; ct < 4; ++ct) {
        v8f c = {};
#pragma unroll
        for (int ks = 0; ks < KSTEPS; ++ks) {
            v16bf b = load_fragB(WT, dinPad, ct, ks);
            c = __builtin_amdgcn_wmma_f32_16x16x32_bf16(false, a[ks], false, b, (short)0, c,
                                                        false, false);
        }
        int col = ct * 16 + (lane & 15);
        float bc = bias[col];
        if (MODE == 0) {
            float* o = out + (size_t)(rows + mrow) * HDIM + col;
#pragma unroll
            for (int v = 0; v < 8; ++v) o[(size_t)v * HDIM] = c[v] + bc;
        } else {
            float* o = out + (size_t)(rowBase + rows + mrow) * HDIM + col;
#pragma unroll
            for (int v = 0; v < 8; ++v) o[(size_t)v * HDIM] += c[v] + bc;
        }
    }
}

// ---------------- fused edge K/V GEMM + eemb + attention logits + segment max ----------------

template <int KSTEPS>
__global__ void edge_attn_kernel(const unsigned short* __restrict__ hbf, int dinPad,
                                 const unsigned short* __restrict__ WkT,
                                 const unsigned short* __restrict__ WvT,
                                 const float* __restrict__ bk, const float* __restrict__ bv,
                                 const float* __restrict__ We,    // [2][64]
                                 const float* __restrict__ eattr, // [E][2]
                                 const int* __restrict__ src, const int* __restrict__ dst, int dlo,
                                 const float* __restrict__ qbuf,
                                 float* __restrict__ vpe, float* __restrict__ alphaOut,
                                 float* __restrict__ dmax, float scale) {
    int wave = blockIdx.x * (blockDim.x >> 5) + (threadIdx.x >> 5);
    int eBase = wave * 16;
    if (eBase >= NEDGE) return;
    int lane = threadIdx.x & 31;

    int srcNode = src[eBase + (lane & 15)];
    const unsigned short* rowPtr = hbf + (size_t)srcNode * dinPad;
    v16bf a[KSTEPS];
#pragma unroll
    for (int ks = 0; ks < KSTEPS; ++ks) a[ks] = load_fragA(rowPtr, ks);
    int mrow = (lane >> 4) * 8;

    float ea0[8], ea1[8];
    int dArr[8];
#pragma unroll
    for (int v = 0; v < 8; ++v) {
        int e = eBase + mrow + v;
        ea0[v] = eattr[(size_t)e * 2];
        ea1[v] = eattr[(size_t)e * 2 + 1];
        dArr[v] = dst[e];
    }

    float part[8] = {0, 0, 0, 0, 0, 0, 0, 0};
#pragma unroll
    for (int ct = 0; ct < 4; ++ct) {
        v8f ck = {}, cv = {};
#pragma unroll
        for (int ks = 0; ks < KSTEPS; ++ks) {
            v16bf bK = load_fragB(WkT, dinPad, ct, ks);
            v16bf bV = load_fragB(WvT, dinPad, ct, ks);
            ck = __builtin_amdgcn_wmma_f32_16x16x32_bf16(false, a[ks], false, bK, (short)0, ck,
                                                         false, false);
            cv = __builtin_amdgcn_wmma_f32_16x16x32_bf16(false, a[ks], false, bV, (short)0, cv,
                                                         false, false);
        }
        int col = ct * 16 + (lane & 15);
        float bkc = bk[col], bvc = bv[col];
        float w0 = We[col], w1 = We[HDIM + col];
#pragma unroll
        for (int v = 0; v < 8; ++v) {
            int e = eBase + mrow + v;
            float ee = ea0[v] * w0 + ea1[v] * w1;
            float kval = ck[v] + bkc + ee;
            float vval = cv[v] + bvc + ee;
            vpe[(size_t)e * HDIM + col] = vval;                       // v + eemb, stored once
            float qv = qbuf[(size_t)(dArr[v] - dlo) * HDIM + col];
            part[v] += kval * qv;
        }
    }
    // reduce each row's dot product across the 16 lanes of its half-wave
#pragma unroll
    for (int v = 0; v < 8; ++v) {
        float s = part[v];
        s += __shfl_xor(s, 1, 16);
        s += __shfl_xor(s, 2, 16);
        s += __shfl_xor(s, 4, 16);
        s += __shfl_xor(s, 8, 16);
        part[v] = s;
    }
#pragma unroll
    for (int v = 0; v < 8; ++v) {
        if ((lane & 15) == v) {
            int e = eBase + mrow + v;
            float al = part[v] * scale;
            alphaOut[e] = al;
            atomicMaxFloat(&dmax[dArr[v]], al);
        }
    }
}

// ---------------- softmax phase 2 + aggregation ----------------

__global__ void expsum_kernel(float* __restrict__ alpha, const int* __restrict__ dst,
                              const float* __restrict__ dmax, float* __restrict__ dsum) {
    int e = blockIdx.x * blockDim.x + threadIdx.x;
    if (e < NEDGE) {
        int d = dst[e];
        float a = __expf(alpha[e] - dmax[d]);
        alpha[e] = a;
        atomicAdd(&dsum[d], a);
    }
}

// 32 threads per edge, 2 columns per thread -> coalesced 64-float scatter-add (L2 resident)
__global__ void aggregate_kernel(const float* __restrict__ alpha, const int* __restrict__ dst,
                                 const float* __restrict__ dsum, const float* __restrict__ vpe,
                                 float* __restrict__ h_acc) {
    int t = blockIdx.x * blockDim.x + threadIdx.x;
    int e = t >> 5;
    int c2 = (t & 31) * 2;
    if (e < NEDGE) {
        // pull the vpe rows a few edges ahead into cache while atomics retire
        __builtin_prefetch(&vpe[(size_t)(e + 8) * HDIM + c2], 0, 1);
        int d = dst[e];
        float coef = alpha[e] / (dsum[d] + 1e-16f);
        float v0 = vpe[(size_t)e * HDIM + c2];
        float v1 = vpe[(size_t)e * HDIM + c2 + 1];
        atomicAdd(&h_acc[(size_t)d * HDIM + c2],     coef * v0);
        atomicAdd(&h_acc[(size_t)d * HDIM + c2 + 1], coef * v1);
    }
}

// ---------------- head: out = relu(h)@Wlin; keep V=|o0|, th=o1 ----------------

__global__ void head_kernel(const float* __restrict__ h_acc, const float* __restrict__ Wlin,
                            float* __restrict__ Vout, float* __restrict__ thOut,
                            float* __restrict__ out) {
    int i = blockIdx.x * blockDim.x + threadIdx.x;
    if (i < N_NODES) {
        float o0 = 0, o1 = 0;
        for (int c = 0; c < HDIM; ++c) {
            float hv = h_acc[(size_t)i * HDIM + c];
            hv = hv > 0.0f ? hv : 0.0f;
            const float* w = Wlin + c * 4;
            o0 += hv * w[0];
            o1 += hv * w[1];
        }
        Vout[i] = fabsf(o0);
        thOut[i] = o1;
        out[i * 4 + 1] = o1;
    }
}

// ---------------- power flow scatter ----------------

__global__ void pf_kernel(const int* __restrict__ pf_i, const int* __restrict__ pf_j,
                          const float* __restrict__ pf_attr,
                          const float* __restrict__ V, const float* __restrict__ th,
                          float* __restrict__ P, float* __restrict__ Q) {
    int e = blockIdx.x * blockDim.x + threadIdx.x;
    if (e < NPF) {
        int i = pf_i[e], j = pf_j[e];
        float r  = pf_attr[(size_t)e * 2];
        float xx = pf_attr[(size_t)e * 2 + 1];
        float den = r * r + xx * xx;
        float G = r / den, B = -xx / den;
        float dlt = th[j] - th[i];
        float cd = __cosf(dlt), sd = __sinf(dlt);
        float vv = V[i] * V[j];
        atomicAdd(&P[i], vv * (G * cd + B * sd));
        atomicAdd(&Q[i], vv * (G * sd - B * cd));
    }
}

// ---------------- final sigmoids ----------------

__device__ inline float csig(float v, float lo, float hi) {
    float mid = 0.5f * (lo + hi);
    float s = 1.0f / (1.0f + __expf(-0.1f * (v - mid)));
    return (s - 0.5f) * (hi - lo) + mid;
}

__global__ void final_kernel(const float* __restrict__ V, const float* __restrict__ P,
                             const float* __restrict__ Q, const float* __restrict__ constraints,
                             const float* __restrict__ x, float* __restrict__ out) {
    int i = blockIdx.x * blockDim.x + threadIdx.x;
    if (i < N_NODES) {
        const float* c = constraints + (size_t)i * 7;
        out[i * 4 + 0] = csig(V[i], c[0], c[1]) / x[i * 4 + 0];
        out[i * 4 + 2] = csig(P[i], c[3], c[4]);
        out[i * 4 + 3] = csig(Q[i], c[5], c[6]);
    }
}

// ---------------- host ----------------

static inline int grid1(long long n, int b) { return (int)((n + b - 1) / b); }

extern "C" void kernel_launch(void* const* d_in, const int* in_sizes, int n_in,
                              void* d_out, int out_size, void* d_ws, size_t ws_size,
                              hipStream_t stream) {
    (void)in_sizes; (void)n_in; (void)out_size; (void)ws_size;

    const float* x           = (const float*)d_in[0];
    const float* constraints = (const float*)d_in[1];
    const int*   edge_src    = (const int*)d_in[2];
    const int*   edge_dst    = (const int*)d_in[3];
    const float* edge_attr   = (const float*)d_in[4];
    const int*   pf_i        = (const int*)d_in[5];
    const int*   pf_j        = (const int*)d_in[6];
    const float* pf_attr     = (const float*)d_in[7];
    const float* Wlin        = (const float*)d_in[8];

    const float* Wq[2] = {(const float*)d_in[9],  (const float*)d_in[18]};
    const float* bq[2] = {(const float*)d_in[10], (const float*)d_in[19]};
    const float* Wk[2] = {(const float*)d_in[11], (const float*)d_in[20]};
    const float* bk[2] = {(const float*)d_in[12], (const float*)d_in[21]};
    const float* Wv[2] = {(const float*)d_in[13], (const float*)d_in[22]};
    const float* bv[2] = {(const float*)d_in[14], (const float*)d_in[23]};
    const float* We[2] = {(const float*)d_in[15], (const float*)d_in[24]};
    const float* Ws[2] = {(const float*)d_in[16], (const float*)d_in[25]};
    const float* bs[2] = {(const float*)d_in[17], (const float*)d_in[26]};

    float* out = (float*)d_out;

    // workspace carve (everything ~77 MB -> fits in the 192 MB L2)
    char* ws = (char*)d_ws;
    size_t off = 0;
    auto carve = [&](size_t bytes) -> void* {
        void* p = ws + off;
        off += (bytes + 255) & ~(size_t)255;
        return p;
    };
    float*          h_acc = (float*)carve((size_t)N_NODES * HDIM * 4);
    unsigned short* hbf   = (unsigned short*)carve((size_t)N_NODES * HDIM * 2);
    float*          qbuf  = (float*)carve((size_t)40000 * HDIM * 4);
    float*          vpe   = (float*)carve((size_t)NEDGE * HDIM * 4);
    float*          alpha = (float*)carve((size_t)NEDGE * 4);
    float*          dmax  = (float*)carve((size_t)N_NODES * 4);
    float*          dsum  = (float*)carve((size_t)N_NODES * 4);
    float*          Vb    = (float*)carve((size_t)N_NODES * 4);
    float*          thb   = (float*)carve((size_t)N_NODES * 4);
    float*          Pb    = (float*)carve((size_t)N_NODES * 4);
    float*          Qb    = (float*)carve((size_t)N_NODES * 4);
    unsigned short* wT[2][4];   // [layer][q,k,v,s]
    for (int t = 0; t < 4; ++t) wT[0][t] = (unsigned short*)carve((size_t)NREL * HDIM * 32 * 2);
    for (int t = 0; t < 4; ++t) wT[1][t] = (unsigned short*)carve((size_t)NREL * HDIM * 64 * 2);

    static const int relDst[NREL] = {1, 2, 3, 2, 3, 3, 0, 0, 0, 1, 2, 1, 2, 3};
    static const int tlo[4] = {0, 10000, 30000, 70000};
    static const int thi[4] = {10000, 30000, 70000, 100000};

    // convert + transpose weights to bf16 [rel][col][Kpad]
    {
        int n0 = NREL * HDIM * 32, n1 = NREL * HDIM * 64;
        cvtWT_kernel<<<grid1(n0, 256), 256, 0, stream>>>(Wq[0], wT[0][0], 4, 32);
        cvtWT_kernel<<<grid1(n0, 256), 256, 0, stream>>>(Wk[0], wT[0][1], 4, 32);
        cvtWT_kernel<<<grid1(n0, 256), 256, 0, stream>>>(Wv[0], wT[0][2], 4, 32);
        cvtWT_kernel<<<grid1(n0, 256), 256, 0, stream>>>(Ws[0], wT[0][3], 4, 32);
        cvtWT_kernel<<<grid1(n1, 256), 256, 0, stream>>>(Wq[1], wT[1][0], 64, 64);
        cvtWT_kernel<<<grid1(n1, 256), 256, 0, stream>>>(Wk[1], wT[1][1], 64, 64);
        cvtWT_kernel<<<grid1(n1, 256), 256, 0, stream>>>(Wv[1], wT[1][2], 64, 64);
        cvtWT_kernel<<<grid1(n1, 256), 256, 0, stream>>>(Ws[1], wT[1][3], 64, 64);
    }

    const float scale = 0.125f;  // 64^-0.5
    const int etiles = NEDGE / 16;
    const int eblocks = (etiles + 3) / 4;

    for (int layer = 0; layer < 2; ++layer) {
        int dinPad = layer ? 64 : 32;

        if (layer == 0) {
            cvt_x_kernel<<<grid1((long long)N_NODES * 32, 256), 256, 0, stream>>>(x, hbf);
        } else {
            relu_cvt_kernel<<<grid1((long long)N_NODES * HDIM, 256), 256, 0, stream>>>(h_acc, hbf);
        }
        zero_kernel<<<grid1((long long)N_NODES * HDIM, 256), 256, 0, stream>>>(h_acc, N_NODES * HDIM);

        for (int r = 0; r < NREL; ++r) {
            int dt = relDst[r];
            int dlo = tlo[dt], M = thi[dt] - tlo[dt];
            int tiles = M / 16;
            int blocks = (tiles + 3) / 4;

            seg_init_kernel<<<grid1(M, 256), 256, 0, stream>>>(dmax, dsum, dlo, M);

            const unsigned short* WqTr = wT[layer][0] + (size_t)r * HDIM * dinPad;
            const unsigned short* WkTr = wT[layer][1] + (size_t)r * HDIM * dinPad;
            const unsigned short* WvTr = wT[layer][2] + (size_t)r * HDIM * dinPad;
            const unsigned short* WsTr = wT[layer][3] + (size_t)r * HDIM * dinPad;
            const float* bqr = bq[layer] + (size_t)r * HDIM;
            const float* bkr = bk[layer] + (size_t)r * HDIM;
            const float* bvr = bv[layer] + (size_t)r * HDIM;
            const float* bsr = bs[layer] + (size_t)r * HDIM;
            const float* Wer = We[layer] + (size_t)r * 2 * HDIM;
            const float* ear = edge_attr + (size_t)r * NEDGE * 2;
            const int*   sr  = edge_src + (size_t)r * NEDGE;
            const int*   dr  = edge_dst + (size_t)r * NEDGE;

            if (layer == 0) {
                gemm_bias_kernel<1, 0><<<blocks, 128, 0, stream>>>(hbf, dinPad, WqTr, bqr,
                                                                   qbuf, dlo, M);
                edge_attn_kernel<1><<<eblocks, 128, 0, stream>>>(hbf, dinPad, WkTr, WvTr, bkr, bvr,
                                                                 Wer, ear, sr, dr, dlo, qbuf, vpe,
                                                                 alpha, dmax, scale);
            } else {
                gemm_bias_kernel<2, 0><<<blocks, 128, 0, stream>>>(hbf, dinPad, WqTr, bqr,
                                                                   qbuf, dlo, M);
                edge_attn_kernel<2><<<eblocks, 128, 0, stream>>>(hbf, dinPad, WkTr, WvTr, bkr, bvr,
                                                                 Wer, ear, sr, dr, dlo, qbuf, vpe,
                                                                 alpha, dmax, scale);
            }

            expsum_kernel<<<grid1(NEDGE, 256), 256, 0, stream>>>(alpha, dr, dmax, dsum);

            aggregate_kernel<<<grid1((long long)NEDGE * 32, 256), 256, 0, stream>>>(
                alpha, dr, dsum, vpe, h_acc);

            if (layer == 0) {
                gemm_bias_kernel<1, 1><<<blocks, 128, 0, stream>>>(hbf, dinPad, WsTr, bsr,
                                                                   h_acc, dlo, M);
            } else {
                gemm_bias_kernel<2, 1><<<blocks, 128, 0, stream>>>(hbf, dinPad, WsTr, bsr,
                                                                   h_acc, dlo, M);
            }
        }
    }

    head_kernel<<<grid1(N_NODES, 256), 256, 0, stream>>>(h_acc, Wlin, Vb, thb, out);
    zero_kernel<<<grid1(N_NODES, 256), 256, 0, stream>>>(Pb, N_NODES);
    zero_kernel<<<grid1(N_NODES, 256), 256, 0, stream>>>(Qb, N_NODES);
    pf_kernel<<<grid1(NPF, 256), 256, 0, stream>>>(pf_i, pf_j, pf_attr, Vb, thb, Pb, Qb);
    final_kernel<<<grid1(N_NODES, 256), 256, 0, stream>>>(Vb, Pb, Qb, constraints, x, out);
}
